// ToyNet_13202729468426
// MI455X (gfx1250) — 8-bit matrix operand order repaired and checked by reading, compiled, not run on hardware
//
#include <hip/hip_runtime.h>


#define BATCH 512
#define KDIM  8192
#define NCLS  8192
#define FLATW 150528

#define MBLK        32
#define LDS_STRIDE  8208
#define LDS_BYTES   (MBLK * LDS_STRIDE)

typedef __attribute__((ext_vector_type(16))) int   v16i;
typedef __attribute__((ext_vector_type(8)))  float v8f;

#if __has_builtin(__builtin_amdgcn_global_load_async_to_lds_b128)
#define HAVE_ASYNC_LDS 1
typedef __attribute__((__vector_size__(4 * sizeof(int)))) int async_v4i;
typedef __attribute__((address_space(1))) async_v4i g1_v4i;
typedef __attribute__((address_space(3))) async_v4i g3_v4i;
#else
#define HAVE_ASYNC_LDS 0
#endif

__device__ __forceinline__ unsigned sign_fp8_byte(float v) {
    return (v > 0.0f) ? 0x38u : ((v < 0.0f) ? 0xB8u : 0x00u);
}

__global__ void bnn_quant_x_fp8(const float* __restrict__ in,
                                unsigned char* __restrict__ xq) {
    size_t t    = (size_t)blockIdx.x * blockDim.x + threadIdx.x;
    size_t idx4 = t * 4;
    size_t b    = idx4 / KDIM;
    size_t i    = idx4 % KDIM;
    const float* p = in + b * FLATW + i;
    unsigned r = sign_fp8_byte(p[0])
               | (sign_fp8_byte(p[1]) << 8)
               | (sign_fp8_byte(p[2]) << 16)
               | (sign_fp8_byte(p[3]) << 24);
    ((unsigned*)xq)[t] = r;
}

__global__ void bnn_quant_w_fp8(const float* __restrict__ W,
                                unsigned char* __restrict__ wq) {
    size_t t = (size_t)blockIdx.x * blockDim.x + threadIdx.x;
    const float* p = W + t * 4;
    unsigned r = sign_fp8_byte(p[0])
               | (sign_fp8_byte(p[1]) << 8)
               | (sign_fp8_byte(p[2]) << 16)
               | (sign_fp8_byte(p[3]) << 24);
    ((unsigned*)wq)[t] = r;
}

__device__ __forceinline__ float sign_m2(float a) {
    return (a > 2.0f) ? 1.0f : ((a < 2.0f) ? -1.0f : 0.0f);
}

__device__ __forceinline__ v16i load_A16(const unsigned char* p) {
    const int2* pa = (const int2*)p;
    int2 a0 = pa[0],  a1 = pa[2],  a2 = pa[4],  a3 = pa[6];
    int2 a4 = pa[8],  a5 = pa[10], a6 = pa[12], a7 = pa[14];
    v16i A = { a0.x, a0.y, a1.x, a1.y, a2.x, a2.y, a3.x, a3.y,
               a4.x, a4.y, a5.x, a5.y, a6.x, a6.y, a7.x, a7.y };
    return A;
}

__device__ __forceinline__ v16i load_B16(const unsigned char* p) {
    const int2* pb = (const int2*)p;
    int2 c0 = pb[0], c1 = pb[2], c2 = pb[4], c3 = pb[6];
    int2 c4 = pb[8], c5 = pb[10], c6 = pb[12], c7 = pb[14];
    v16i B = { c0.x,c0.y, c1.x,c1.y, c2.x,c2.y, c3.x,c3.y,
               c4.x,c4.y, c5.x,c5.y, c6.x,c6.y, c7.x,c7.y };
    return B;
}

__global__ __launch_bounds__(256)
void bnn_wmma_fp8(const unsigned char* __restrict__ xq,
                  const unsigned char* __restrict__ wq,
                  float* __restrict__ out) {
    extern __shared__ unsigned char smem[];

    const int lane = threadIdx.x & 31;
    const int wv   = threadIdx.x >> 5;
    const int half = lane >> 4;
    const int l15  = lane & 15;

    const int mBase = blockIdx.y * MBLK;
    const int nBase = blockIdx.x * 512 + wv * 64;

    for (int c = threadIdx.x; c < MBLK * (KDIM / 16); c += 256) {
        const int row = c >> 9;
        const int col = (c & 511) * 16;
        const unsigned char* src = xq + (size_t)(mBase + row) * KDIM + col;
        const unsigned       dst = (unsigned)(row * LDS_STRIDE + col);
#if HAVE_ASYNC_LDS
        __builtin_amdgcn_global_load_async_to_lds_b128(
            (g1_v4i*)(uintptr_t)src,
            (g3_v4i*)(unsigned)(uintptr_t)(smem + dst), 0, 0);
#else
        *(int4*)(smem + dst) = *(const int4*)src;
#endif
    }
#if HAVE_ASYNC_LDS
#if __has_builtin(__builtin_amdgcn_s_wait_asynccnt)
    __builtin_amdgcn_s_wait_asynccnt(0);
#else
    asm volatile("s_wait_asynccnt 0x0" ::: "memory");
#endif
#endif
    __syncthreads();

    const unsigned char* aP0 = smem + (unsigned)((0  + l15) * LDS_STRIDE + half * 8);
    const unsigned char* aP1 = smem + (unsigned)((16 + l15) * LDS_STRIDE + half * 8);

    const unsigned char* bP0 = wq + (size_t)(nBase +  0 + l15) * KDIM + half * 8;
    const unsigned char* bP1 = wq + (size_t)(nBase + 16 + l15) * KDIM + half * 8;
    const unsigned char* bP2 = wq + (size_t)(nBase + 32 + l15) * KDIM + half * 8;
    const unsigned char* bP3 = wq + (size_t)(nBase + 48 + l15) * KDIM + half * 8;

    v8f acc00 = {}, acc01 = {}, acc02 = {}, acc03 = {};
    v8f acc10 = {}, acc11 = {}, acc12 = {}, acc13 = {};

    for (int k = 0; k < KDIM; k += 128) {
        v16i A0 = load_A16(aP0 + k);
        v16i A1 = load_A16(aP1 + k);

        v16i B = load_B16(bP0 + k);
        acc00 = __builtin_amdgcn_wmma_f32_16x16x128_fp8_fp8(A0, B, (short)0, acc00, false, false);
        acc10 = __builtin_amdgcn_wmma_f32_16x16x128_fp8_fp8(A1, B, (short)0, acc10, false, false);

        B = load_B16(bP1 + k);
        acc01 = __builtin_amdgcn_wmma_f32_16x16x128_fp8_fp8(A0, B, (short)0, acc01, false, false);
        acc11 = __builtin_amdgcn_wmma_f32_16x16x128_fp8_fp8(A1, B, (short)0, acc11, false, false);

        B = load_B16(bP2 + k);
        acc02 = __builtin_amdgcn_wmma_f32_16x16x128_fp8_fp8(A0, B, (short)0, acc02, false, false);
        acc12 = __builtin_amdgcn_wmma_f32_16x16x128_fp8_fp8(A1, B, (short)0, acc12, false, false);

        B = load_B16(bP3 + k);
        acc03 = __builtin_amdgcn_wmma_f32_16x16x128_fp8_fp8(A0, B, (short)0, acc03, false, false);
        acc13 = __builtin_amdgcn_wmma_f32_16x16x128_fp8_fp8(A1, B, (short)0, acc13, false, false);
    }

    {
        const int mRow = mBase + half * 8;
        float* o = out + (size_t)mRow * NCLS + nBase + l15;
#pragma unroll
        for (int r = 0; r < 8; ++r) {
            o[(size_t)r * NCLS +  0] = sign_m2(acc00[r]);
            o[(size_t)r * NCLS + 16] = sign_m2(acc01[r]);
            o[(size_t)r * NCLS + 32] = sign_m2(acc02[r]);
            o[(size_t)r * NCLS + 48] = sign_m2(acc03[r]);
        }
        float* o1 = o + (size_t)16 * NCLS;
#pragma unroll
        for (int r = 0; r < 8; ++r) {
            o1[(size_t)r * NCLS +  0] = sign_m2(acc10[r]);
            o1[(size_t)r * NCLS + 16] = sign_m2(acc11[r]);
            o1[(size_t)r * NCLS + 32] = sign_m2(acc12[r]);
            o1[(size_t)r * NCLS + 48] = sign_m2(acc13[r]);
        }
    }
}

extern "C" void kernel_launch(void* const* d_in, const int* in_sizes, int n_in,
                              void* d_out, int out_size, void* d_ws, size_t ws_size,
                              hipStream_t stream) {
    const float* input = (const float*)d_in[0];
    const float* W     = (const float*)d_in[1];
    float*       out   = (float*)d_out;

    unsigned char* xq = (unsigned char*)d_ws;
    unsigned char* wq = xq + (size_t)BATCH * KDIM;

    bnn_quant_x_fp8<<<(BATCH * KDIM / 4) / 256, 256, 0, stream>>>(input, xq);
    bnn_quant_w_fp8<<<((size_t)NCLS * KDIM / 4) / 256, 256, 0, stream>>>(W, wq);

    dim3 grid(NCLS / 512, BATCH / MBLK);
    bnn_wmma_fp8<<<grid, 256, LDS_BYTES, stream>>>(xq, wq, out);
}
